// PathAttention_59837484367886
// MI455X (gfx1250) — matrix operand order repaired and checked by reading, not compiled, not run on hardware
//
#include <hip/hip_runtime.h>
#include <hip/hip_bf16.h>

// Problem dims (fixed by the reference): x[B,P,E], w1[H,E], w2[1,H]
// B=256, P=512, E=256, H=128.  M = B*P = 131072 fused rows.
static constexpr int kP = 512;
static constexpr int kE = 256;   // K dim of GEMM1
static constexpr int kH = 128;   // N dim of GEMM1

typedef __attribute__((ext_vector_type(16))) _Float16 v16h;
typedef __attribute__((ext_vector_type(8)))  _Float16 v8h;
typedef __attribute__((ext_vector_type(8)))  float    v8f;

// ---------------------------------------------------------------------------
// Kernel 0: convert w1 (f32 [H,E]) -> f16 for the WMMA path (tiny, one-shot)
// ---------------------------------------------------------------------------
__global__ void w1_to_f16_kernel(const float* __restrict__ w1,
                                 _Float16* __restrict__ w1h, int n) {
    int i = blockIdx.x * blockDim.x + threadIdx.x;
    if (i < n) w1h[i] = (_Float16)w1[i];
}

// ---------------------------------------------------------------------------
// Kernel 1: scores[m] = sum_n tanh( (x @ w1^T)[m,n] ) * w2[n]
// One workgroup = one 16-row M tile. 8 waves, wave w owns N tile [16w,16w+16).
// v_wmma_f32_16x16x32_f16, K=256 in 8 unrolled steps.
// ---------------------------------------------------------------------------
__global__ __launch_bounds__(256) void scores_kernel(
    const float* __restrict__ x,        // [M, E] f32 (M = B*P)
    const _Float16* __restrict__ w1h,   // [H, E] f16
    const float* __restrict__ w2,       // [H] f32
    float* __restrict__ scores)         // [M] f32
{
    __shared__ __align__(16) _Float16 ldsA[16 * kE];  // 16 x 256 f16 A tile (8 KB)
    __shared__ float ldsS[16];                        // per-M score accumulators

    const int tid  = threadIdx.x;
    const int lane = tid & 31;
    const int wave = tid >> 5;                 // 0..7 -> N tile
    const size_t tile = blockIdx.x;            // M tile index

    // ---- Stage A tile into LDS, f32 -> f16 (each thread: 16 contiguous elems)
    {
        const float* xg = x + tile * (size_t)(16 * kE) + (size_t)tid * 16;
        v8h h0, h1;
        #pragma unroll
        for (int i = 0; i < 8; ++i) h0[i] = (_Float16)xg[i];
        #pragma unroll
        for (int i = 0; i < 8; ++i) h1[i] = (_Float16)xg[8 + i];
        *reinterpret_cast<v8h*>(&ldsA[tid * 16])     = h0;
        *reinterpret_cast<v8h*>(&ldsA[tid * 16 + 8]) = h1;
    }
    if (tid < 16) ldsS[tid] = 0.0f;
    __syncthreads();

    // ---- WMMA fragment addressing (ISA 7.12.2, 16-bit A 16x32 / B 32x16):
    // A: lanes 0-15 = row M=lane, K {k0..k0+7, k0+16..k0+23};
    //    lanes 16-31 = row M=lane-16, K {k0+8..k0+15, k0+24..k0+31}.
    // B: lanes 0-15 = col N=lane, K {k0..k0+7, k0+16..k0+23}; lanes 16-31 = col N, K {k0+8..k0+15, k0+24..k0+31}.
    const int half = lane >> 4;
    const int rowA = lane & 15;
    const int nloc = lane & 15;
    const size_t wrow = (size_t)(wave * 16 + nloc) * kE;  // w1 row = output feature

    v8f acc = {};
    #pragma unroll
    for (int k0 = 0; k0 < kE; k0 += 32) {
        const int ka = k0 + (half ? 8 : 0);
        v8h a_lo = *reinterpret_cast<const v8h*>(&ldsA[rowA * kE + ka]);
        v8h a_hi = *reinterpret_cast<const v8h*>(&ldsA[rowA * kE + ka + 16]);
        v16h a = __builtin_shufflevector(a_lo, a_hi,
                 0,1,2,3,4,5,6,7,8,9,10,11,12,13,14,15);

        const int kb = k0 + (half ? 8 : 0);
        const _Float16* wp = w1h + wrow + kb;
        v8h b_lo = *reinterpret_cast<const v8h*>(wp);
        v8h b_hi = *reinterpret_cast<const v8h*>(wp + 16);
        v16h b = __builtin_shufflevector(b_lo, b_hi,
                 0,1,2,3,4,5,6,7,8,9,10,11,12,13,14,15);

        acc = __builtin_amdgcn_wmma_f32_16x16x32_f16(
                  false, a, false, b, (short)0, acc, false, false);
    }

    // ---- tanh + dot with w2 slice, reduce across the 16 lanes of each half.
    // C/D layout: VGPR r holds M = r + 8*half, lanes within half hold N = lane&15.
    const float w2v = w2[wave * 16 + nloc];
    float s[8];
    #pragma unroll
    for (int r = 0; r < 8; ++r) s[r] = tanhf(acc[r]) * w2v;
    #pragma unroll
    for (int m = 8; m >= 1; m >>= 1) {
        #pragma unroll
        for (int r = 0; r < 8; ++r) s[r] += __shfl_xor(s[r], m, 32);
    }
    if ((lane & 15) == 0) {               // lane 0 -> M=r, lane 16 -> M=r+8
        #pragma unroll
        for (int r = 0; r < 8; ++r) atomicAdd(&ldsS[r + 8 * half], s[r]);
    }
    __syncthreads();
    if (tid < 16) scores[tile * 16 + tid] = ldsS[tid];
}

// ---------------------------------------------------------------------------
// Kernel 2: per batch b: softmax over P, then pooled[b,:] = sum_p w[p]*x[b,p,:]
// One workgroup per b; thread tid owns output column e = tid.
// ---------------------------------------------------------------------------
__global__ __launch_bounds__(256) void softmax_pool_kernel(
    const float* __restrict__ x,       // [B, P, E]
    const float* __restrict__ scores,  // [B, P]
    float* __restrict__ out)           // [B, E]
{
    __shared__ float red[256];
    __shared__ float wts[kP];
    const int tid = threadIdx.x;
    const int b = blockIdx.x;

    const float s0 = scores[b * kP + tid];
    const float s1 = scores[b * kP + 256 + tid];

    // max over P
    red[tid] = fmaxf(s0, s1);
    __syncthreads();
    for (int off = 128; off > 0; off >>= 1) {
        if (tid < off) red[tid] = fmaxf(red[tid], red[tid + off]);
        __syncthreads();
    }
    const float mx = red[0];
    __syncthreads();

    // exp + sum
    const float e0 = __expf(s0 - mx);
    const float e1 = __expf(s1 - mx);
    red[tid] = e0 + e1;
    __syncthreads();
    for (int off = 128; off > 0; off >>= 1) {
        if (tid < off) red[tid] += red[tid + off];
        __syncthreads();
    }
    const float inv = 1.0f / red[0];
    wts[tid]       = e0 * inv;
    wts[tid + 256] = e1 * inv;
    __syncthreads();

    // weighted pool over P (coalesced: 1 KB per workgroup per p)
    const float* xb = x + (size_t)b * kP * kE;
    float acc = 0.0f;
    for (int p0 = 0; p0 < kP; p0 += 8) {
        __builtin_prefetch(xb + (size_t)(p0 + 24) * kE + tid, 0, 0);  // global_prefetch_b8
        #pragma unroll
        for (int j = 0; j < 8; ++j)
            acc = fmaf(wts[p0 + j], xb[(size_t)(p0 + j) * kE + tid], acc);
    }
    out[b * kE + tid] = acc;
}

// ---------------------------------------------------------------------------
extern "C" void kernel_launch(void* const* d_in, const int* in_sizes, int n_in,
                              void* d_out, int out_size, void* d_ws, size_t ws_size,
                              hipStream_t stream) {
    (void)in_sizes; (void)n_in; (void)out_size; (void)ws_size;
    const float* x  = (const float*)d_in[0];   // [256,512,256] f32
    const float* w1 = (const float*)d_in[1];   // [128,256] f32
    const float* w2 = (const float*)d_in[2];   // [1,128] f32
    float* out = (float*)d_out;                // [256,256] f32

    // workspace: [0,64KB) w1 in f16; [64KB, 64KB+512KB) scores f32 [B*P]
    _Float16* w1h  = (_Float16*)d_ws;
    float* scores  = (float*)((char*)d_ws + 64 * 1024);

    w1_to_f16_kernel<<<(kH * kE) / 256, 256, 0, stream>>>(w1, w1h, kH * kE);
    scores_kernel<<<(256 * kP) / 16, 256, 0, stream>>>(x, w1h, w2, scores);
    softmax_pool_kernel<<<256, 256, 0, stream>>>(x, scores, out);
}
